// ScaledDotProductAttention_44289702756791
// MI455X (gfx1250) — matrix operand order repaired and checked by reading, not compiled, not run on hardware
//
#include <hip/hip_runtime.h>

typedef __attribute__((ext_vector_type(16))) _Float16 v16h;
typedef __attribute__((ext_vector_type(2)))  _Float16 h2;
typedef __attribute__((ext_vector_type(2)))  __fp16   f16x2;
typedef __attribute__((ext_vector_type(8)))  float    v8f;
typedef __attribute__((ext_vector_type(4)))  float    v4f;
typedef __attribute__((ext_vector_type(4)))  int      v4i;

union AFrag {
    v16h     v;
    _Float16 h[16];
    h2       p[8];
    v4i      q[2];
};
union Row8 {
    _Float16 h[8];
    h2       p[4];
    v4i      q;
};

// v_cvt_pk_f16_f32 (RTZ pack), bit-cast to _Float16x2
static __device__ __forceinline__ h2 pk16(float a, float b) {
    f16x2 t = __builtin_amdgcn_cvt_pkrtz(a, b);
    return __builtin_bit_cast(h2, t);
}

#define WMMA_F16(A, B, C) \
    __builtin_amdgcn_wmma_f32_16x16x32_f16(false, (A), false, (B), (short)0, (C), false, false)

// Flash attention for [B, 2048, 64] x [B, 2048, 64], int32 mask, fp32 in/out.
// One block = 128 To rows (8 waves x 16). Scores computed transposed
// (S^T = K_tile * Q^T) so P becomes the PV A-fragment by in-lane f16 packing.
// K/V staged double-buffered in LDS as f16; mask/output use non-temporal
// hints so the streamed 268 MB mask cannot evict the L2-resident K/V (67 MB).
__launch_bounds__(256)
__global__ void fa_wmma_kernel(const float* __restrict__ q,
                               const float* __restrict__ k,
                               const float* __restrict__ v,
                               const int*   __restrict__ mask,
                               float*       __restrict__ out,
                               int To, int Ti)
{
    __shared__ _Float16 sK[2][32][64];   // [buf][ti_row][d]   row-major, 2x4 KB
    __shared__ _Float16 sV[2][64][32];   // [buf][d][ti_row]   transposed, 2x4 KB

    const int tid  = threadIdx.x;
    const int lane = tid & 31;
    const int wave = tid >> 5;
    const int half = lane >> 4;          // 0: lanes 0-15, 1: lanes 16-31
    const int ln   = lane & 15;

    const int tilesPerB = To >> 7;
    const int b         = blockIdx.x / tilesPerB;
    const int tile      = blockIdx.x % tilesPerB;
    const int to_base   = tile * 128 + wave * 16;

    // ---- Q as two B-matrix fragments (K = d 0..31 and 32..63), scale baked in ----
    AFrag qb[2];
    {
        const float* qp = q + ((size_t)(b * To + to_base + ln) << 6);
#pragma unroll
        for (int j = 0; j < 2; ++j) {
            const float* p = qp + j * 32 + half * 8;
#pragma unroll
            for (int e = 0; e < 8; ++e)
                qb[j].p[e] = pk16(p[2 * e + ((e & 4) << 1)] * 0.125f, p[2 * e + ((e & 4) << 1) + 1] * 0.125f);
        }
    }

    v8f o0 = {}, o1 = {}, o2 = {}, o3 = {};      // O: 16 To x 64 d (4 N-tiles)
    float m_run = -3.0e38f;                      // per-lane stats, row = ln
    float l_run = 0.0f;

    const int* mrow = mask + (size_t)(b * To + to_base + ln) * Ti;

    // ---- staging thread mappings ----
    const int rK = tid >> 3,        cK = (tid & 7) << 3;    // K: 1 row x 8 cols
    const int rV = (tid >> 4) << 1, cV = (tid & 15) << 2;   // V: 2 rows x 4 cols
    const float* kbase = k + (((size_t)b * Ti + rK) << 6) + cK;
    const float* vbase = v + (((size_t)b * Ti + rV) << 6) + cV;

    // prefetch tile 0
    v4f kr0 = *(const v4f*)kbase;
    v4f kr1 = *(const v4f*)(kbase + 4);
    v4f vr0 = *(const v4f*)vbase;
    v4f vr1 = *(const v4f*)(vbase + 64);

    const int nIter = Ti >> 5;
    for (int it = 0; it < nIter; ++it) {
        const int buf = it & 1;

        // ---- store staged registers to LDS (f16, V transposed) ----
        {
            Row8 kh;
            kh.p[0] = pk16(kr0[0], kr0[1]);
            kh.p[1] = pk16(kr0[2], kr0[3]);
            kh.p[2] = pk16(kr1[0], kr1[1]);
            kh.p[3] = pk16(kr1[2], kr1[3]);
            *(v4i*)&sK[buf][rK][cK] = kh.q;                        // ds_store_b128
#pragma unroll
            for (int e = 0; e < 4; ++e)                            // 4x ds_store_b32
                *(h2*)&sV[buf][cV + e][rV] = pk16(vr0[e], vr1[e]);
        }
        // ---- prefetch next tile into registers (overlaps with compute below) ----
        if (it + 1 < nIter) {
            const size_t off = ((size_t)(it + 1)) << 11;           // 32 rows * 64
            kr0 = *(const v4f*)(kbase + off);
            kr1 = *(const v4f*)(kbase + off + 4);
            vr0 = *(const v4f*)(vbase + off);
            vr1 = *(const v4f*)(vbase + off + 64);
        }
        __syncthreads();

        const int ti0 = it << 5;

        // ---- streamed mask loads (non-temporal: read-once, keep K/V in L2) ----
        const int* mp = mrow + ti0 + half * 8;
        const v4i mA0 = __builtin_nontemporal_load((const v4i*)mp);
        const v4i mA1 = __builtin_nontemporal_load((const v4i*)(mp + 4));
        const v4i mB0 = __builtin_nontemporal_load((const v4i*)(mp + 16));
        const v4i mB1 = __builtin_nontemporal_load((const v4i*)(mp + 20));

        // ---- S^T = K_tile x Q^T for the two 16-row Ti subtiles ----
        float p0[8], p1[8];
#pragma unroll
        for (int sub = 0; sub < 2; ++sub) {
            AFrag a0, a1;
            const int krow = sub * 16 + ln;
            const int dofs = half * 8;
            a0.q[0] = *(const v4i*)&sK[buf][krow][dofs];
            a0.q[1] = *(const v4i*)&sK[buf][krow][dofs + 16];
            a1.q[0] = *(const v4i*)&sK[buf][krow][32 + dofs];
            a1.q[1] = *(const v4i*)&sK[buf][krow][32 + dofs + 16];
            v8f s = {};
            s = WMMA_F16(a0.v, qb[0].v, s);
            s = WMMA_F16(a1.v, qb[1].v, s);
            const v4i m0 = sub ? mB0 : mA0;
            const v4i m1 = sub ? mB1 : mA1;
            float* ps = sub ? p1 : p0;
#pragma unroll
            for (int r = 0; r < 4; ++r) ps[r]     = (m0[r] == 0) ? -1.0e10f : s[r];
#pragma unroll
            for (int r = 0; r < 4; ++r) ps[4 + r] = (m1[r] == 0) ? -1.0e10f : s[4 + r];
        }

        // ---- online softmax (row = lane&15) ----
        float tmax = p0[0];
#pragma unroll
        for (int r = 1; r < 8; ++r) tmax = fmaxf(tmax, p0[r]);
#pragma unroll
        for (int r = 0; r < 8; ++r) tmax = fmaxf(tmax, p1[r]);
        tmax = fmaxf(tmax, __shfl_xor(tmax, 16, 32));
        const float m_new = fmaxf(m_run, tmax);
        const float alpha = __expf(m_run - m_new);
        m_run = m_new;

        float e0[8], e1[8];
        float lsum = 0.0f;
#pragma unroll
        for (int r = 0; r < 8; ++r) { e0[r] = __expf(p0[r] - m_new); lsum += e0[r]; }
#pragma unroll
        for (int r = 0; r < 8; ++r) { e1[r] = __expf(p1[r] - m_new); lsum += e1[r]; }
        lsum += __shfl_xor(lsum, 16, 32);
        l_run = l_run * alpha + lsum;

        AFrag pa;                                    // P as 16x32 A-fragment
#pragma unroll
        for (int j = 0; j < 4; ++j) {
            pa.p[j]     = pk16(e0[2 * j], e0[2 * j + 1]);
            pa.p[4 + j] = pk16(e1[2 * j], e1[2 * j + 1]);
        }

        // ---- rescale accumulator: row of O vgpr r is r + half*8 ----
#pragma unroll
        for (int r = 0; r < 8; ++r) {
            const float ar = __shfl(alpha, r + ((lane & 16) >> 1), 32);
            o0[r] *= ar; o1[r] *= ar; o2[r] *= ar; o3[r] *= ar;
        }

        // ---- O += P x V (4 d-tiles) ----
        {
            const int kofs = half * 8;
            AFrag bv;
            bv.q[0] = *(const v4i*)&sV[buf][ln][kofs];
            bv.q[1] = *(const v4i*)&sV[buf][ln][kofs + 16];
            o0 = WMMA_F16(pa.v, bv.v, o0);
            bv.q[0] = *(const v4i*)&sV[buf][16 + ln][kofs];
            bv.q[1] = *(const v4i*)&sV[buf][16 + ln][kofs + 16];
            o1 = WMMA_F16(pa.v, bv.v, o1);
            bv.q[0] = *(const v4i*)&sV[buf][32 + ln][kofs];
            bv.q[1] = *(const v4i*)&sV[buf][32 + ln][kofs + 16];
            o2 = WMMA_F16(pa.v, bv.v, o2);
            bv.q[0] = *(const v4i*)&sV[buf][48 + ln][kofs];
            bv.q[1] = *(const v4i*)&sV[buf][48 + ln][kofs + 16];
            o3 = WMMA_F16(pa.v, bv.v, o3);
        }
    }

    // ---- epilogue: O / l, non-temporal stores (write-once) ----
    const float inv = 1.0f / l_run;
    const size_t orow_base = (size_t)(b * To + to_base);
#pragma unroll
    for (int r = 0; r < 8; ++r) {
        const float sinv = __shfl(inv, r + ((lane & 16) >> 1), 32);
        float* op = out + ((orow_base + r + half * 8) << 6) + ln;
        __builtin_nontemporal_store(o0[r] * sinv, op);
        __builtin_nontemporal_store(o1[r] * sinv, op + 16);
        __builtin_nontemporal_store(o2[r] * sinv, op + 32);
        __builtin_nontemporal_store(o3[r] * sinv, op + 48);
    }
}

extern "C" void kernel_launch(void* const* d_in, const int* in_sizes, int n_in,
                              void* d_out, int out_size, void* d_ws, size_t ws_size,
                              hipStream_t stream) {
    const float* q    = (const float*)d_in[0];
    const float* k    = (const float*)d_in[1];
    const float* v    = (const float*)d_in[2];
    const int*   mask = (const int*)d_in[3];
    float*       out  = (float*)d_out;

    const long BTo = (long)in_sizes[0] / 64;   // B * To
    const long BTi = (long)in_sizes[1] / 64;   // B * Ti
    const long MT  = (long)in_sizes[3];        // B * To * Ti
    const int  Ti  = (int)(MT / BTo);
    const int  To  = (int)(MT / BTi);

    const int blocks = (int)(BTo / 128);       // 128 To-rows per block
    fa_wmma_kernel<<<blocks, 256, 0, stream>>>(q, k, v, mask, out, To, Ti);
}
